// FullyContextualEmbedding_13297218748595
// MI455X (gfx1250) — matrix operand order repaired and checked by reading, not compiled, not run on hardware
//
#include <hip/hip_runtime.h>
#include <cstdint>
#include <cstddef>

// ---------- problem constants ----------
#define FEAT   1024
#define NQ     4096
#define KSUP   64
#define XDX    4096   // X width: [f | r | h0 | h1]
#define XDW    3072   // Wcat width: [W_ih (2048) | W_hh (1024)]

typedef __bf16 bf16;
typedef __attribute__((ext_vector_type(4)))  __bf16 bf16x4;
typedef __attribute__((ext_vector_type(8)))  __bf16 bf16x8;
typedef __attribute__((ext_vector_type(16))) __bf16 bf16x16;
typedef __attribute__((ext_vector_type(4)))  float  f32x4;
typedef __attribute__((ext_vector_type(8)))  float  f32x8;

// ---------- WMMA helpers (CDNA5 16x16x32 bf16, fp32 accum) ----------
static __device__ inline bf16x16 fce_cat16(bf16x8 lo, bf16x8 hi) {
    return __builtin_shufflevector(lo, hi, 0,1,2,3,4,5,6,7,8,9,10,11,12,13,14,15);
}

// A-fragment (16x32 bf16): lane L holds row M=L%16; half=L/16.
// VGPR 0..7 -> K = half*16 + [0..15] (contiguous).
static __device__ inline bf16x16 fce_ldA(const bf16* row_kc, int half) {
    bf16x8 lo = *(const bf16x8*)(row_kc + half * 16);
    bf16x8 hi = *(const bf16x8*)(row_kc + half * 16 + 8);
    return fce_cat16(lo, hi);
}

// B-fragment (32x16 bf16): lane L holds col N=L%16; half=L/16.
// K = half*16 .. half*16+15 is 32 CONTIGUOUS bytes -> single vector load
// (two adjacent global_load_b128, no register shuffling).
static __device__ inline bf16x16 fce_ldB(const bf16* nrow_kc, int half) {
    return *(const bf16x16*)(nrow_kc + half * 16);
}

static __device__ inline f32x8 fce_wmma(bf16x16 a, bf16x16 b, f32x8 c) {
    return __builtin_amdgcn_wmma_f32_16x16x32_bf16(false, a, false, b,
                                                   (short)0, c, false, false);
}

static __device__ inline float fce_sigmoid(float x) {
    return 1.0f / (1.0f + __expf(-x));
}

// ---------- one-time prep kernels ----------
__global__ void __launch_bounds__(256)
fce_conv_w(const float* __restrict__ Wih, const float* __restrict__ Whh,
           bf16* __restrict__ Wcat) {
    size_t e = ((size_t)blockIdx.x * 256 + threadIdx.x) * 8; // 4096*3072 elems
    int j = (int)(e / XDW);
    int d = (int)(e % XDW);
    const float* src = (d < 2 * FEAT) ? (Wih + (size_t)j * (2 * FEAT) + d)
                                      : (Whh + (size_t)j * FEAT + (d - 2 * FEAT));
    f32x4 v0 = *(const f32x4*)(src);
    f32x4 v1 = *(const f32x4*)(src + 4);
    bf16x8 o;
#pragma unroll
    for (int i = 0; i < 4; ++i) { o[i] = (bf16)v0[i]; o[i + 4] = (bf16)v1[i]; }
    *(bf16x8*)(Wcat + e) = o;
}

__global__ void __launch_bounds__(256)
fce_conv_g(const float* __restrict__ G, bf16* __restrict__ Gb,
           bf16* __restrict__ Gt) {
    size_t e = ((size_t)blockIdx.x * 256 + threadIdx.x) * 4; // 64*1024 elems
    int k = (int)(e >> 10);
    int d = (int)(e & 1023);
    f32x4 v = *(const f32x4*)(G + e);
    bf16x4 o;
#pragma unroll
    for (int i = 0; i < 4; ++i) o[i] = (bf16)v[i];
    *(bf16x4*)(Gb + e) = o;                      // [k][d] row-major
#pragma unroll
    for (int i = 0; i < 4; ++i)
        Gt[(size_t)(d + i) * KSUP + k] = o[i];   // [d][k] transposed
}

__global__ void __launch_bounds__(256)
fce_bsum(const float* __restrict__ bih, const float* __restrict__ bhh,
         float* __restrict__ bsum) {
    int j = blockIdx.x * 256 + threadIdx.x; // 4096
    bsum[j] = bih[j] + bhh[j];
}

__global__ void __launch_bounds__(256)
fce_init_x(const float* __restrict__ f, bf16* __restrict__ X,
           float* __restrict__ c) {
    size_t e = ((size_t)blockIdx.x * 256 + threadIdx.x) * 4; // 4096*1024 elems
    int n = (int)(e >> 10);
    int u = (int)(e & 1023);
    f32x4 v = *(const f32x4*)(f + e);
    bf16x4 o;
#pragma unroll
    for (int i = 0; i < 4; ++i) o[i] = (bf16)v[i];
    *(bf16x4*)(X + (size_t)n * XDX + u) = o;            // f block (constant)
    *(bf16x4*)(X + (size_t)n * XDX + 2 * FEAT + u) = o; // h0 = f
    f32x4 z = {0.f, 0.f, 0.f, 0.f};
    *(f32x4*)(c + e) = z;
}

// ---------- per-step kernel 1: attention (logits -> softmax -> r) ----------
// 32 blocks x 256 threads; wave w handles 16 query rows. hoff = h-read block.
__global__ void __launch_bounds__(256)
fce_attn(bf16* __restrict__ X, const bf16* __restrict__ Gb,
         const bf16* __restrict__ Gt, int hoff) {
    __shared__ __align__(16) bf16 aLds[8][16][72]; // stride 72 -> 16B aligned rows
    int lane  = threadIdx.x & 31;
    int wave  = threadIdx.x >> 5;
    int laneM = lane & 15;
    int half  = lane >> 4;
    int rbase = blockIdx.x * 128 + wave * 16;

    // logits = h @ G^T : 16x64 via 4 N-tiles, K=1024.
    // Fully unrolled + double-buffered: no backedge -> no register rotation.
    const bf16* hrow = X + (size_t)(rbase + laneM) * XDX + hoff;
    const bf16* Grow[4];
#pragma unroll
    for (int t = 0; t < 4; ++t) Grow[t] = Gb + (size_t)(t * 16 + laneM) * FEAT;

    f32x8 acc[4] = {};
    bf16x16 a0, a1, bb0[4], bb1[4];
    a0 = fce_ldA(hrow, half);
#pragma unroll
    for (int t = 0; t < 4; ++t) bb0[t] = fce_ldB(Grow[t], half);

#pragma unroll
    for (int kc = 0; kc < FEAT; kc += 64) {
        int k1 = kc + 32;
        a1 = fce_ldA(hrow + k1, half);
#pragma unroll
        for (int t = 0; t < 4; ++t) bb1[t] = fce_ldB(Grow[t] + k1, half);
#pragma unroll
        for (int t = 0; t < 4; ++t) acc[t] = fce_wmma(a0, bb0[t], acc[t]);
        int k2 = (kc + 64 < FEAT) ? kc + 64 : 0;
        a0 = fce_ldA(hrow + k2, half);
#pragma unroll
        for (int t = 0; t < 4; ++t) bb0[t] = fce_ldB(Grow[t] + k2, half);
#pragma unroll
        for (int t = 0; t < 4; ++t) acc[t] = fce_wmma(a1, bb1[t], acc[t]);
    }

    // softmax over 64 per row; row M=r+8*half lives in 16-lane half-groups.
#pragma unroll
    for (int r = 0; r < 8; ++r) {
        float m = fmaxf(fmaxf(acc[0][r], acc[1][r]), fmaxf(acc[2][r], acc[3][r]));
#pragma unroll
        for (int msk = 8; msk >= 1; msk >>= 1) m = fmaxf(m, __shfl_xor(m, msk, 32));
        float e0 = __expf(acc[0][r] - m), e1 = __expf(acc[1][r] - m);
        float e2 = __expf(acc[2][r] - m), e3 = __expf(acc[3][r] - m);
        float s = e0 + e1 + e2 + e3;
#pragma unroll
        for (int msk = 8; msk >= 1; msk >>= 1) s += __shfl_xor(s, msk, 32);
        float inv = 1.0f / s;
        int row = r + 8 * half;
        aLds[wave][row][0 * 16 + laneM] = (bf16)(e0 * inv);
        aLds[wave][row][1 * 16 + laneM] = (bf16)(e1 * inv);
        aLds[wave][row][2 * 16 + laneM] = (bf16)(e2 * inv);
        aLds[wave][row][3 * 16 + laneM] = (bf16)(e3 * inv);
    }
    // re-layout a into A-fragments via LDS (intra-wave only; DS is in-order)
    const bf16* arow = &aLds[wave][laneM][0];
    bf16x16 aa0 = fce_ldA(arow, half);       // K = 0..31
    bf16x16 aa1 = fce_ldA(arow + 32, half);  // K = 32..63

    // r = a @ G : 64 d-tiles; stage 4 tiles in LDS, write out as b128.
    for (int dg = 0; dg < 16; ++dg) {
#pragma unroll
        for (int dl = 0; dl < 4; ++dl) {
            int dt = dg * 4 + dl;
            const bf16* gt = Gt + (size_t)(dt * 16 + laneM) * KSUP;
            f32x8 racc = {};
            racc = fce_wmma(aa0, fce_ldB(gt, half), racc);
            racc = fce_wmma(aa1, fce_ldB(gt + 32, half), racc);
#pragma unroll
            for (int r = 0; r < 8; ++r)
                aLds[wave][r + 8 * half][dl * 16 + laneM] = (bf16)racc[r];
        }
        // coalesced write-out: 16 rows x 64 cols bf16 -> b128 stores
#pragma unroll
        for (int j = 0; j < 4; ++j) {
            int id  = j * 32 + lane;
            int row = id >> 3;
            int ch  = id & 7;
            bf16x8 v = *(const bf16x8*)(&aLds[wave][row][ch * 8]);
            *(bf16x8*)(X + (size_t)(rbase + row) * XDX + FEAT + dg * 64 + ch * 8) = v;
        }
    }
}

// ---------- per-step kernel 2: fused gates GEMM + LSTM update ----------
// Wave tile: 64 rows x 16 hidden units, accumulating all FOUR gate blocks
// (N = g*1024 + u) -> 4x4 accumulators, 16 WMMA : 8 b128-load pairs per chunk.
// Epilogue applies the LSTM nonlinearity in-register: no gates buffer at all.
// grid (FEAT/32, NQ/256) x 256 threads; WG tile 256 rows x 32 units.
__global__ void __launch_bounds__(256)
fce_gemm_lstm(bf16* __restrict__ X, const bf16* __restrict__ Wc,
              const float* __restrict__ bsum, const float* __restrict__ fsrc,
              float* __restrict__ c, float* __restrict__ hout,
              int hoff, int hwoff) {
    int lane  = threadIdx.x & 31;
    int wave  = threadIdx.x >> 5;
    int laneM = lane & 15;
    int half  = lane >> 4;
    int Mb = blockIdx.y * 256 + (wave >> 1) * 64;
    int Ub = blockIdx.x * 32  + (wave & 1) * 16;

    const bf16* Arow[4];
    const bf16* Brow[4];
#pragma unroll
    for (int i = 0; i < 4; ++i)
        Arow[i] = X + (size_t)(Mb + i * 16 + laneM) * XDX;
#pragma unroll
    for (int g = 0; g < 4; ++g)
        Brow[g] = Wc + (size_t)(g * FEAT + Ub + laneM) * XDW;

    // K mapping: Wcat k 0..2047 <-> X [f|r] 0..2047 ; k 2048..3071 <-> X h-read
    f32x8 acc[4][4] = {};
    bf16x16 a0[4], b0[4], a1[4], b1[4];

#pragma unroll
    for (int i = 0; i < 4; ++i) {
        a0[i] = fce_ldA(Arow[i], half);
        b0[i] = fce_ldB(Brow[i], half);
    }

#pragma unroll 4
    for (int kc = 0; kc < XDW; kc += 64) {
        int k1  = kc + 32;
        int k1m = (k1 < 2 * FEAT) ? k1 : (k1 - 2 * FEAT + hoff);
#pragma unroll
        for (int i = 0; i < 4; ++i) {
            a1[i] = fce_ldA(Arow[i] + k1m, half);
            b1[i] = fce_ldB(Brow[i] + k1, half);
        }
#pragma unroll
        for (int mi = 0; mi < 4; ++mi)
#pragma unroll
            for (int g = 0; g < 4; ++g)
                acc[mi][g] = fce_wmma(a0[mi], b0[g], acc[mi][g]);
        int k2  = (kc + 64 < XDW) ? kc + 64 : 0;
        int k2m = (k2 < 2 * FEAT) ? k2 : (k2 - 2 * FEAT + hoff);
#pragma unroll
        for (int i = 0; i < 4; ++i) {
            a0[i] = fce_ldA(Arow[i] + k2m, half);
            b0[i] = fce_ldB(Brow[i] + k2, half);
        }
#pragma unroll
        for (int mi = 0; mi < 4; ++mi)
#pragma unroll
            for (int g = 0; g < 4; ++g)
                acc[mi][g] = fce_wmma(a1[mi], b1[g], acc[mi][g]);
    }

    // epilogue: LSTM nonlinearity, c/h update
    int col = Ub + laneM;
    float bs0 = bsum[0 * FEAT + col];
    float bs1 = bsum[1 * FEAT + col];
    float bs2 = bsum[2 * FEAT + col];
    float bs3 = bsum[3 * FEAT + col];
#pragma unroll
    for (int mi = 0; mi < 4; ++mi) {
#pragma unroll
        for (int r = 0; r < 8; ++r) {
            int row = Mb + mi * 16 + r + 8 * half;
            size_t be = (size_t)row * FEAT + col;
            float gi = acc[mi][0][r] + bs0;
            float gf = acc[mi][1][r] + bs1;
            float gg = acc[mi][2][r] + bs2;
            float go = acc[mi][3][r] + bs3;
            float cn = fce_sigmoid(gf) * c[be] + fce_sigmoid(gi) * tanhf(gg);
            float hn = fce_sigmoid(go) * tanhf(cn) + fsrc[be];
            c[be]    = cn;
            hout[be] = hn;
            X[(size_t)row * XDX + hwoff + col] = (bf16)hn;
        }
    }
}

// ---------- launcher ----------
extern "C" void kernel_launch(void* const* d_in, const int* in_sizes, int n_in,
                              void* d_out, int out_size, void* d_ws, size_t ws_size,
                              hipStream_t stream) {
    const float* f    = (const float*)d_in[0];
    const float* G    = (const float*)d_in[1];
    const float* W_ih = (const float*)d_in[2];
    const float* W_hh = (const float*)d_in[3];
    const float* b_ih = (const float*)d_in[4];
    const float* b_hh = (const float*)d_in[5];
    float* hout = (float*)d_out;

    // workspace layout (bytes, 256-aligned)
    const size_t OFF_WCAT = 0;                                     // 4096*3072 bf16
    const size_t OFF_X    = OFF_WCAT + (size_t)4096 * XDW * 2;     // 4096*4096 bf16
    const size_t OFF_GB   = OFF_X    + (size_t)NQ * XDX * 2;       // 64*1024 bf16
    const size_t OFF_GT   = OFF_GB   + (size_t)KSUP * FEAT * 2;    // 1024*64 bf16
    const size_t OFF_BSUM = OFF_GT   + (size_t)FEAT * KSUP * 2;    // 4096 f32
    const size_t OFF_C    = OFF_BSUM + (size_t)4096 * 4;           // 4096*1024 f32
    const size_t NEED     = OFF_C + (size_t)NQ * FEAT * 4;
    if (ws_size < NEED) return;

    char* ws = (char*)d_ws;
    bf16*  Wcat = (bf16*)(ws + OFF_WCAT);
    bf16*  X    = (bf16*)(ws + OFF_X);
    bf16*  Gb   = (bf16*)(ws + OFF_GB);
    bf16*  Gt   = (bf16*)(ws + OFF_GT);
    float* bsum = (float*)(ws + OFF_BSUM);
    float* c    = (float*)(ws + OFF_C);

    fce_conv_w<<<(4096 * XDW) / (256 * 8), 256, 0, stream>>>(W_ih, W_hh, Wcat);
    fce_conv_g<<<(KSUP * FEAT) / (256 * 4), 256, 0, stream>>>(G, Gb, Gt);
    fce_bsum<<<4096 / 256, 256, 0, stream>>>(b_ih, b_hh, bsum);
    fce_init_x<<<(NQ * FEAT) / (256 * 4), 256, 0, stream>>>(f, X, c);

    for (int s = 0; s < KSUP; ++s) {
        int hr = (s & 1) ? 3 * FEAT : 2 * FEAT; // h read block
        int hw = (s & 1) ? 2 * FEAT : 3 * FEAT; // h write block
        fce_attn<<<NQ / 128, 256, 0, stream>>>(X, Gb, Gt, hr);
        fce_gemm_lstm<<<dim3(FEAT / 32, NQ / 256), 256, 0, stream>>>(
            X, Wcat, bsum, f, c, hout, hr, hw);
    }
}
